// HomoGraphSAGEEncoder_27754078667399
// MI455X (gfx1250) — matrix operand order repaired and checked by reading, not compiled, not run on hardware
//
#include <hip/hip_runtime.h>

typedef unsigned short u16;
typedef __attribute__((ext_vector_type(16))) __bf16 v16bf;
typedef __attribute__((ext_vector_type(8)))  float  v8f;

#define N_NODES 100000
#define N_EDGES 1600000
#define D 128
#define NF (N_NODES * D)          // 12,800,000 elements
#define LDS_STRIDE 136            // 128 bf16 + 8 pad -> 272B row stride (kills bank conflicts)

union Frag16 { uint4 q[2]; v16bf v; };

__device__ __forceinline__ u16 f32_to_bf16(float f) {
    unsigned u = __float_as_uint(f);
    unsigned r = u + 0x7FFFu + ((u >> 16) & 1u);   // round-to-nearest-even
    return (u16)(r >> 16);
}

// ---------------- utility kernels ----------------

__global__ __launch_bounds__(256) void zero_f32(float* __restrict__ p, int n) {
    int i = blockIdx.x * 256 + threadIdx.x;
    if (i < n) p[i] = 0.0f;
}

__global__ __launch_bounds__(256) void degree_kernel(const int* __restrict__ dst,
                                                     float* __restrict__ deg) {
    int e = blockIdx.x * 256 + threadIdx.x;
    if (e < N_EDGES) atomicAdd(&deg[dst[e]], 1.0f);
}

// W (k-major, 128x128 f32) -> Wt (col-major bf16): Wt[n*128 + k] = bf16(W[k*128 + n])
__global__ __launch_bounds__(256) void transpose_cvt_w(const float* __restrict__ W,
                                                       u16* __restrict__ Wt) {
    int i = blockIdx.x * 256 + threadIdx.x;       // 16384 threads
    if (i < D * D) {
        int k = i >> 7, n = i & 127;
        Wt[n * D + k] = f32_to_bf16(W[i]);
    }
}

__global__ __launch_bounds__(256) void cvt_bf16_kernel(const float* __restrict__ x,
                                                       u16* __restrict__ o, int n) {
    int i = blockIdx.x * 256 + threadIdx.x;
    if (i < n) o[i] = f32_to_bf16(x[i]);
}

// wave-per-edge scatter: lane l adds h[src][4l..4l+3] into agg[dst][4l..4l+3]
__global__ __launch_bounds__(256) void scatter_add_kernel(const float* __restrict__ h,
                                                          const int* __restrict__ src,
                                                          const int* __restrict__ dst,
                                                          float* __restrict__ agg) {
    int tid  = blockIdx.x * 256 + threadIdx.x;
    int e    = tid >> 5;                 // 32 lanes per edge (wave32: uniform per wave)
    int lane = tid & 31;
    if (e < N_EDGES) {
        int s = src[e];
        int d = dst[e];
        const float4* srow = (const float4*)(h + (size_t)s * D);
        float4 v = srow[lane];           // 512B coalesced gather per wave
        float* drow = agg + (size_t)d * D + lane * 4;
        atomicAdd(&drow[0], v.x);
        atomicAdd(&drow[1], v.y);
        atomicAdd(&drow[2], v.z);
        atomicAdd(&drow[3], v.w);
    }
}

// aggb = bf16(agg / max(deg,1))
__global__ __launch_bounds__(256) void div_cvt_kernel(const float* __restrict__ agg,
                                                      const float* __restrict__ deg,
                                                      u16* __restrict__ aggb) {
    int i = blockIdx.x * 256 + threadIdx.x;
    if (i < NF) {
        float d = fmaxf(deg[i >> 7], 1.0f);
        aggb[i] = f32_to_bf16(agg[i] / d);
    }
}

// ---------------- fused SAGE GEMM ----------------
// out[16 rows x 128 cols] = hb@WsT^T + aggb@WnT^T + bias  (optionally ReLU)
// block = 256 thr = 8 waves; wave w owns col tile [16w,16w+16); 6250 blocks cover all rows.
__global__ __launch_bounds__(256) void sage_gemm_kernel(const u16* __restrict__ hb,
                                                        const u16* __restrict__ aggb,
                                                        const u16* __restrict__ WsT,
                                                        const u16* __restrict__ WnT,
                                                        const float* __restrict__ bias,
                                                        float* __restrict__ out,
                                                        int relu) {
    __shared__ u16 lAs[16 * LDS_STRIDE];
    __shared__ u16 lAn[16 * LDS_STRIDE];

    const int t     = threadIdx.x;
    const int node0 = blockIdx.x * 16;

    // Stage both 16x128 bf16 A-tiles: each thread copies one 16B chunk per tile.
    {
        int r = t >> 4, c = t & 15;                  // c = 16B chunk (8 bf16)
        const uint4* ss = (const uint4*)(hb   + (size_t)(node0 + r) * D);
        const uint4* sn = (const uint4*)(aggb + (size_t)(node0 + r) * D);
        *(uint4*)(&lAs[r * LDS_STRIDE + c * 8]) = ss[c];
        *(uint4*)(&lAn[r * LDS_STRIDE + c * 8]) = sn[c];
    }
    __syncthreads();

    const int wave = t >> 5;
    const int lane = t & 31;
    const int half = lane >> 4;        // 0 | 1
    const int mrow = lane & 15;        // A row held by this lane
    const int col  = lane & 15;        // B/C column within tile
    const int gcol = wave * 16 + col;  // global output column

    v8f acc = {};
    #pragma unroll
    for (int kk = 0; kk < D; kk += 32) {
        // A fragment (16x32 bf16): lane<16 -> K {kk..kk+7, kk+16..kk+23}; lane>=16 -> +8
        Frag16 aS, aN;
        const uint4* ps = (const uint4*)(&lAs[mrow * LDS_STRIDE + kk + half * 8]);
        const uint4* pn = (const uint4*)(&lAn[mrow * LDS_STRIDE + kk + half * 8]);
        aS.q[0] = ps[0]; aS.q[1] = ps[2];            // +32B = K+16 group
        aN.q[0] = pn[0]; aN.q[1] = pn[2];

        // B fragment (32x16 bf16): lane holds column `col`, K {kk..kk+7, kk+16..kk+23}; lane>=16 -> +8
        Frag16 bS, bN;
        const uint4* qs = (const uint4*)(WsT + (size_t)gcol * D + kk + half * 8);
        const uint4* qn = (const uint4*)(WnT + (size_t)gcol * D + kk + half * 8);
        bS.q[0] = qs[0]; bS.q[1] = qs[2];
        bN.q[0] = qn[0]; bN.q[1] = qn[2];

        acc = __builtin_amdgcn_wmma_f32_16x16x32_bf16(false, aN.v, false, bN.v,
                                                      (short)0, acc, false, false);
        acc = __builtin_amdgcn_wmma_f32_16x16x32_bf16(false, aS.v, false, bS.v,
                                                      (short)0, acc, false, false);
    }

    // C/D layout: VGPR i -> M = i + half*8, N = col
    const float bv = bias[gcol];
    #pragma unroll
    for (int i = 0; i < 8; ++i) {
        int m = half * 8 + i;
        float v = acc[i] + bv;
        if (relu) v = fmaxf(v, 0.0f);
        out[(size_t)(node0 + m) * D + gcol] = v;
    }
}

// ---------------- host launch ----------------

extern "C" void kernel_launch(void* const* d_in, const int* in_sizes, int n_in,
                              void* d_out, int out_size, void* d_ws, size_t ws_size,
                              hipStream_t stream) {
    const float* x       = (const float*)d_in[0];
    const float* W_self0 = (const float*)d_in[1];
    const float* b0      = (const float*)d_in[2];
    const float* W_neigh0= (const float*)d_in[3];
    const float* W_self1 = (const float*)d_in[4];
    const float* b1      = (const float*)d_in[5];
    const float* W_neigh1= (const float*)d_in[6];
    const int*   src     = (const int*)d_in[7];
    const int*   dst     = (const int*)d_in[8];

    // workspace carve (~154 MB)
    char* w = (char*)d_ws;
    float* deg  = (float*)w;  w += (size_t)102400 * 4;
    float* agg  = (float*)w;  w += (size_t)NF * 4;
    float* h1   = (float*)w;  w += (size_t)NF * 4;
    u16*   hb   = (u16*)w;    w += (size_t)NF * 2;
    u16*   aggb = (u16*)w;    w += (size_t)NF * 2;
    u16*   Ws0  = (u16*)w;    w += (size_t)D * D * 2;
    u16*   Wn0  = (u16*)w;    w += (size_t)D * D * 2;
    u16*   Ws1  = (u16*)w;    w += (size_t)D * D * 2;
    u16*   Wn1  = (u16*)w;    w += (size_t)D * D * 2;

    const int B256      = 256;
    const int gNodesElt = (NF + 255) / 256;          // 50000
    const int gEdges    = (N_EDGES + 255) / 256;     // 6250
    const int gScatter  = (N_EDGES * 32) / 256;      // 200000
    const int gGemm     = N_NODES / 16;              // 6250 (exact)

    // degrees (shared by both layers)
    zero_f32<<<(N_NODES + 255) / 256, B256, 0, stream>>>(deg, N_NODES);
    degree_kernel<<<gEdges, B256, 0, stream>>>(dst, deg);

    // weights -> transposed bf16
    transpose_cvt_w<<<64, B256, 0, stream>>>(W_self0,  Ws0);
    transpose_cvt_w<<<64, B256, 0, stream>>>(W_neigh0, Wn0);
    transpose_cvt_w<<<64, B256, 0, stream>>>(W_self1,  Ws1);
    transpose_cvt_w<<<64, B256, 0, stream>>>(W_neigh1, Wn1);

    // ---- layer 0 ----
    cvt_bf16_kernel<<<gNodesElt, B256, 0, stream>>>(x, hb, NF);
    zero_f32<<<gNodesElt, B256, 0, stream>>>(agg, NF);
    scatter_add_kernel<<<gScatter, B256, 0, stream>>>(x, src, dst, agg);
    div_cvt_kernel<<<gNodesElt, B256, 0, stream>>>(agg, deg, aggb);
    sage_gemm_kernel<<<gGemm, B256, 0, stream>>>(hb, aggb, Ws0, Wn0, b0, h1, 1);

    // ---- layer 1 ----
    cvt_bf16_kernel<<<gNodesElt, B256, 0, stream>>>(h1, hb, NF);
    zero_f32<<<gNodesElt, B256, 0, stream>>>(agg, NF);
    scatter_add_kernel<<<gScatter, B256, 0, stream>>>(h1, src, dst, agg);
    div_cvt_kernel<<<gNodesElt, B256, 0, stream>>>(agg, deg, aggb);
    sage_gemm_kernel<<<gGemm, B256, 0, stream>>>(hb, aggb, Ws1, Wn1, b1,
                                                 (float*)d_out, 0);
}
